// TransFormerModel_73169062855354
// MI455X (gfx1250) — matrix operand order repaired and checked by reading, not compiled, not run on hardware
//
#include <hip/hip_runtime.h>

// ---------------------------------------------------------------------------
// CDNA5 (gfx1250) transformer forward pass.
// Core math: v_wmma_f32_16x16x32_bf16 (wave32 WMMA).
// Tile staging: global_load_async_to_lds_b128 (ASYNCcnt) when available.
// ---------------------------------------------------------------------------

typedef __attribute__((ext_vector_type(16))) __bf16 v16bf;
typedef __attribute__((ext_vector_type(8)))  float  v8f;
typedef __attribute__((ext_vector_type(4)))  int    v4i;

__device__ __forceinline__ __bf16 f2bf(float f) { return (__bf16)f; }

#if defined(__gfx1250__) && __has_builtin(__builtin_amdgcn_global_load_async_to_lds_b128)
#define USE_ASYNC_LDS 1
#else
#define USE_ASYNC_LDS 0
#endif

#if USE_ASYNC_LDS
typedef __attribute__((address_space(1))) v4i* gptr_v4i;
typedef __attribute__((address_space(3))) v4i* lptr_v4i;
__device__ __forceinline__ void async_cp_b128(const __bf16* g, __bf16* l) {
  __builtin_amdgcn_global_load_async_to_lds_b128(
      (gptr_v4i)(__attribute__((address_space(1))) void*)g,
      (lptr_v4i)(__attribute__((address_space(3))) void*)l, 0, 0);
}
__device__ __forceinline__ void wait_async0() {
#if __has_builtin(__builtin_amdgcn_s_wait_asynccnt)
  __builtin_amdgcn_s_wait_asynccnt(0);
#else
  asm volatile("s_wait_asynccnt 0" ::: "memory");
#endif
}
#endif

// ------------------------- f32 -> bf16 convert (x4) ------------------------
__global__ void cvt_f32_bf16(const float* __restrict__ in, __bf16* __restrict__ out,
                             size_t n) {
  size_t i = ((size_t)blockIdx.x * blockDim.x + threadIdx.x) * 4;
  if (i + 3 < n) {
    float4 f = *reinterpret_cast<const float4*>(in + i);
    out[i]     = f2bf(f.x);
    out[i + 1] = f2bf(f.y);
    out[i + 2] = f2bf(f.z);
    out[i + 3] = f2bf(f.w);
  } else {
    for (; i < n; ++i) out[i] = f2bf(in[i]);
  }
}

// ------------------------- embedding gather --------------------------------
__global__ void embed_kernel(const int* __restrict__ idx, const float* __restrict__ tok,
                             const float* __restrict__ pos, __bf16* __restrict__ xb,
                             int T, int D) {
  int row = blockIdx.x;  // b*T + t
  int t = row % T;
  int id = idx[row];
  const float* te = tok + (size_t)id * D;
  const float* pe = pos + (size_t)t * D;
  __bf16* xo = xb + (size_t)row * D;
  for (int d = threadIdx.x; d < D; d += blockDim.x)
    xo[d] = f2bf(te[d] + pe[d]);
}

// ------------------- v transpose: [B,T,(h,s)] -> [B,H,s,T] -----------------
__global__ void transpose_v(const __bf16* __restrict__ v2, __bf16* __restrict__ vT,
                            int T, int D, int H, int HS) {
  __shared__ __bf16 tile[64][72];
  int z = blockIdx.y;  // b*H + h
  int b = z / H, h = z % H;
  int t0 = blockIdx.x * 64;
  int tx = threadIdx.x & 63, ty = threadIdx.x >> 6;  // 64 x 4
  for (int rr = ty; rr < 64; rr += 4)
    tile[rr][tx] = v2[(size_t)(b * T + t0 + rr) * D + h * HS + tx];
  __syncthreads();
  for (int rr = ty; rr < 64; rr += 4)
    vT[((size_t)z * HS + rr) * T + t0 + tx] = tile[tx][rr];
}

// ------------------------- bf16 WMMA GEMM ----------------------------------
// C[m,n] = scale * sum_k A[m,k] * B[n,k]  (+ bias[n])
//   A row-major [M,K] (M must be a multiple of 128; true for all callers)
//   B row-major [N,K] (N-edge handled by clamped loads + guarded stores:
//     in WMMA, B column n feeds only C column n, so duplicate rows are safe)
// Batched over blockIdx.z:  b1 = z/zdiv, b2 = z%zdiv; ptr += b1*off1 + b2*off2
// Block tile 128x64, 8 waves, each wave a 32x32 micro-tile (2x2 WMMA).
template <int OUT_BF16>
__global__ __launch_bounds__(256) void gemm_wmma(
    const __bf16* __restrict__ A, const __bf16* __restrict__ Bm,
    void* __restrict__ Cv, const float* __restrict__ bias,
    int M, int N, int K, int lda, int ldb, int ldc,
    long long offA1, long long offA2, long long offB1, long long offB2,
    long long offC1, long long offC2, int zdiv, float scale) {
  __shared__ __bf16 sA[128][40];  // 32 valid cols + 16B pad (keeps b128 align)
  __shared__ __bf16 sB[64][40];

  int z = blockIdx.z;
  int b1 = z / zdiv, b2 = z % zdiv;
  A  += (long long)b1 * offA1 + (long long)b2 * offA2;
  Bm += (long long)b1 * offB1 + (long long)b2 * offB2;
  size_t coff = (size_t)((long long)b1 * offC1 + (long long)b2 * offC2);

  int rowBase = blockIdx.y * 128;
  int colBase = blockIdx.x * 64;

  int tid  = threadIdx.x;
  int lane = tid & 31, wid = tid >> 5;
  int wm = wid & 3, wn = wid >> 2;  // 4 waves along M, 2 along N
  int l15 = lane & 15;
  int hi  = lane >> 4;              // lane half (0/1)

  v8f acc[2][2] = {};

  // cooperative tile-load assignments (branch-free)
  int ar = tid >> 1, ac  = (tid & 1) * 16;  // A: 128x32, 16 bf16 (32B) / thread
  int bn = tid >> 2, bkc = (tid & 3) * 8;   // B: 64x32,   8 bf16 (16B) / thread
  int mr  = rowBase + ar;                   // no guard: M % 128 == 0
  int nc  = colBase + bn;
  int ncc = nc < N ? nc : N - 1;            // clamp (safe, see header comment)

  const __bf16* gA = A  + (size_t)mr  * lda;
  const __bf16* gB = Bm + (size_t)ncc * ldb;
  __bf16* lA = &sA[ar][ac];
  __bf16* lB = &sB[bn][bkc];

  for (int k0 = 0; k0 < K; k0 += 32) {  // K % 32 == 0 for all callers
    const __bf16* ga = gA + k0 + ac;
    const __bf16* gb = gB + k0 + bkc;
#if USE_ASYNC_LDS
    async_cp_b128(ga, lA);
    async_cp_b128(ga + 8, lA + 8);
    async_cp_b128(gb, lB);
    wait_async0();
#else
    reinterpret_cast<uint4*>(lA)[0] = reinterpret_cast<const uint4*>(ga)[0];
    reinterpret_cast<uint4*>(lA)[1] = reinterpret_cast<const uint4*>(ga)[1];
    reinterpret_cast<uint4*>(lB)[0] = reinterpret_cast<const uint4*>(gb)[0];
#endif
    __syncthreads();

    // A fragment (16x32): lanes 0-15 -> K 0..15, lanes 16-31 -> K 16..31
    v16bf af[2], bfr[2];
    int koffA = hi * 16;
#pragma unroll
    for (int mi = 0; mi < 2; ++mi) {
      const __bf16* ap = &sA[wm * 32 + mi * 16 + l15][0];
#pragma unroll
      for (int e = 0; e < 16; ++e) af[mi][e] = ap[koffA + e];
    }
    // B fragment (32x16): lane = column, lane half selects K 0-15 / 16-31
    int koffB = hi * 16;
#pragma unroll
    for (int ni = 0; ni < 2; ++ni) {
      const __bf16* bp = &sB[wn * 32 + ni * 16 + l15][0];
#pragma unroll
      for (int e = 0; e < 16; ++e) bfr[ni][e] = bp[koffB + e];
    }
#pragma unroll
    for (int mi = 0; mi < 2; ++mi)
#pragma unroll
      for (int ni = 0; ni < 2; ++ni)
        acc[mi][ni] = __builtin_amdgcn_wmma_f32_16x16x32_bf16(
            false, af[mi], false, bfr[ni], (short)0, acc[mi][ni], false, false);
    __syncthreads();
  }

  // C/D layout: col = lane&15, row = r + 8*(lane>=16); rows always < M
#pragma unroll
  for (int mi = 0; mi < 2; ++mi) {
#pragma unroll
    for (int ni = 0; ni < 2; ++ni) {
      int col = colBase + wn * 32 + ni * 16 + l15;
      if (col < N) {
        float bv = bias ? bias[col] : 0.f;
#pragma unroll
        for (int r = 0; r < 8; ++r) {
          int row = rowBase + wm * 32 + mi * 16 + r + hi * 8;
          float v = acc[mi][ni][r] * scale + bv;
          size_t o = coff + (size_t)row * ldc + col;
          if (OUT_BF16) ((__bf16*)Cv)[o] = f2bf(v);
          else          ((float*)Cv)[o]  = v;
        }
      }
    }
  }
}

// --------------- column softmax over axis i, causal mask j<=i --------------
__global__ void col_softmax(const float* __restrict__ wei, __bf16* __restrict__ out, int T) {
  int z = blockIdx.y, j = blockIdx.x;
  const float* col = wei + (size_t)z * T * T + j;
  __bf16* oc = out + (size_t)z * T * T + j;
  __shared__ float red[256];
  int tid = threadIdx.x;

  float mx = -INFINITY;
  for (int i = j + tid; i < T; i += 256) mx = fmaxf(mx, col[(size_t)i * T]);
  red[tid] = mx; __syncthreads();
  for (int s = 128; s > 0; s >>= 1) { if (tid < s) red[tid] = fmaxf(red[tid], red[tid + s]); __syncthreads(); }
  mx = red[0]; __syncthreads();

  float sum = 0.f;
  for (int i = j + tid; i < T; i += 256) sum += expf(col[(size_t)i * T] - mx);
  red[tid] = sum; __syncthreads();
  for (int s = 128; s > 0; s >>= 1) { if (tid < s) red[tid] += red[tid + s]; __syncthreads(); }
  float inv = 1.f / red[0];

  for (int i = tid; i < T; i += 256) {
    float v = (i >= j) ? expf(col[(size_t)i * T] - mx) * inv : 0.f;
    oc[(size_t)i * T] = f2bf(v);
  }
}

// ------------------------- NLL loss over V ---------------------------------
__global__ void zero1(float* p) { *p = 0.f; }

__global__ void loss_kernel(const float* __restrict__ logits, const int* __restrict__ tgt,
                            float* __restrict__ loss, int V, float invN) {
  int row = blockIdx.x;
  const float* lg = logits + (size_t)row * V;
  __shared__ float red[256];
  int tid = threadIdx.x;

  float mx = -INFINITY;
  for (int v = tid; v < V; v += 256) mx = fmaxf(mx, lg[v]);
  red[tid] = mx; __syncthreads();
  for (int s = 128; s > 0; s >>= 1) { if (tid < s) red[tid] = fmaxf(red[tid], red[tid + s]); __syncthreads(); }
  mx = red[0]; __syncthreads();

  float sum = 0.f;
  for (int v = tid; v < V; v += 256) sum += expf(lg[v] - mx);
  red[tid] = sum; __syncthreads();
  for (int s = 128; s > 0; s >>= 1) { if (tid < s) red[tid] += red[tid + s]; __syncthreads(); }

  if (tid == 0) {
    float lp = lg[tgt[row]] - mx - logf(red[0]);
    atomicAdd(loss, -lp * invN);
  }
}

// ---------------------------------------------------------------------------
extern "C" void kernel_launch(void* const* d_in, const int* in_sizes, int n_in,
                              void* d_out, int out_size, void* d_ws, size_t ws_size,
                              hipStream_t stream) {
  const int B = 4, T = 1024, D = 1024, H = 16, HS = 64, V = 50257;
  const int BT = B * T;

  const int*   idx = (const int*)d_in[0];
  const int*   tgt = (const int*)d_in[1];
  const float* tok = (const float*)d_in[2];
  const float* pos = (const float*)d_in[3];
  const float* Wk  = (const float*)d_in[4];
  const float* Wq  = (const float*)d_in[5];
  const float* Wv  = (const float*)d_in[6];
  const float* lmW = (const float*)d_in[7];
  const float* lmb = (const float*)d_in[8];

  float* logits = (float*)d_out;            // [BT, V] f32
  float* loss   = logits + (size_t)BT * V;  // scalar

  // --- workspace carve ---
  char* p = (char*)d_ws;
  auto carve = [&](size_t bytes) -> char* {
    char* r = p; p += (bytes + 255) & ~(size_t)255; return r;
  };
  __bf16* xb   = (__bf16*)carve((size_t)BT * D * 2);
  __bf16* wkb  = (__bf16*)carve((size_t)D * D * 2);
  __bf16* wqb  = (__bf16*)carve((size_t)D * D * 2);
  __bf16* wvb  = (__bf16*)carve((size_t)D * D * 2);
  __bf16* lmWb = (__bf16*)carve((size_t)V * D * 2);
  __bf16* k2   = (__bf16*)carve((size_t)BT * D * 2);        // [B,T,(h,s)]
  __bf16* q2   = (__bf16*)carve((size_t)BT * D * 2);
  __bf16* v2   = (__bf16*)carve((size_t)BT * D * 2);
  __bf16* vT   = (__bf16*)carve((size_t)BT * D * 2);        // [B,H,s,T]
  float*  weiF = (float*) carve((size_t)B * H * T * T * 4); // raw scores
  __bf16* weiB = (__bf16*)carve((size_t)B * H * T * T * 2); // post-softmax
  __bf16* att  = (__bf16*)carve((size_t)BT * D * 2);        // concat heads

  // --- weight converts (bf16 keeps f32 exponent range) ---
  {
    size_t n = (size_t)D * D;
    unsigned g = (unsigned)((n / 4 + 255) / 256);
    cvt_f32_bf16<<<g, 256, 0, stream>>>(Wk, wkb, n);
    cvt_f32_bf16<<<g, 256, 0, stream>>>(Wq, wqb, n);
    cvt_f32_bf16<<<g, 256, 0, stream>>>(Wv, wvb, n);
    size_t nl = (size_t)V * D;
    cvt_f32_bf16<<<(unsigned)((nl / 4 + 255) / 256), 256, 0, stream>>>(lmW, lmWb, nl);
  }

  // --- embedding ---
  embed_kernel<<<BT, 256, 0, stream>>>(idx, tok, pos, xb, T, D);

  // --- QKV projections: [BT,D] x W[D,D]^T ---
  {
    dim3 g(D / 64, BT / 128, 1);
    gemm_wmma<1><<<g, 256, 0, stream>>>(xb, wkb, k2, nullptr, BT, D, D, D, D, D,
                                        0, 0, 0, 0, 0, 0, 1, 1.f);
    gemm_wmma<1><<<g, 256, 0, stream>>>(xb, wqb, q2, nullptr, BT, D, D, D, D, D,
                                        0, 0, 0, 0, 0, 0, 1, 1.f);
    gemm_wmma<1><<<g, 256, 0, stream>>>(xb, wvb, v2, nullptr, BT, D, D, D, D, D,
                                        0, 0, 0, 0, 0, 0, 1, 1.f);
  }

  // --- v transpose for the PV GEMM ---
  transpose_v<<<dim3(T / 64, B * H), 256, 0, stream>>>(v2, vT, T, D, H, HS);

  // --- scores: wei[b,h,i,j] = (k_i . q_j) / 32 ---
  {
    dim3 g(T / 64, T / 128, B * H);
    gemm_wmma<0><<<g, 256, 0, stream>>>(
        k2, q2, weiF, nullptr, T, T, HS, D, D, T,
        (long long)T * D, (long long)HS,
        (long long)T * D, (long long)HS,
        (long long)H * T * T, (long long)T * T,
        H, 1.f / 32.f);
  }

  // --- masked column softmax (reference softmax axis=2) ---
  col_softmax<<<dim3(T, B * H), 256, 0, stream>>>(weiF, weiB, T);

  // --- out = wei @ v : B = vT[b,h,s,t] (N-major) ---
  {
    dim3 g(HS / 64, T / 128, B * H);
    gemm_wmma<1><<<g, 256, 0, stream>>>(
        weiB, vT, att, nullptr, T, HS, T, T, T, D,
        (long long)H * T * T, (long long)T * T,
        (long long)H * HS * T, (long long)HS * T,
        (long long)T * D, (long long)HS,
        H, 1.f);
  }

  // --- lm_head: logits = att @ lm_W^T + lm_b ---
  {
    dim3 g((V + 63) / 64, BT / 128, 1);
    gemm_wmma<0><<<g, 256, 0, stream>>>(att, lmWb, logits, lmb, BT, V, D, D, D, V,
                                        0, 0, 0, 0, 0, 0, 1, 1.f);
  }

  // --- loss ---
  zero1<<<1, 1, 0, stream>>>(loss);
  loss_kernel<<<BT, 256, 0, stream>>>(logits, tgt, loss, V, 1.f / (float)BT);
}
